// SelfAttention_22376779612294
// MI455X (gfx1250) — matrix operand order repaired and checked by reading, not compiled, not run on hardware
//
#include <hip/hip_runtime.h>

typedef __attribute__((ext_vector_type(16))) _Float16 v16h;
typedef __attribute__((ext_vector_type(8)))  _Float16 v8h;
typedef __attribute__((ext_vector_type(8)))  float    v8f;

__device__ __forceinline__ v8f wmma_f32_f16(v16h a, v16h b, v8f c) {
  // (neg_a, A, neg_b, B, c_mod, C, reuse_a, reuse_b)
  return __builtin_amdgcn_wmma_f32_16x16x32_f16(false, a, false, b, (short)0, c,
                                                false, false);
}

// Convert W[256][ncols] f32 -> Wt[ncols][256] f16 (transposed, contiguous K per column)
__global__ __launch_bounds__(256) void convert_w_kernel(
    const float* __restrict__ W, _Float16* __restrict__ Wt, int ncols)
{
  int i = blockIdx.x * 256 + threadIdx.x;   // over ncols*256 elements
  int c = i >> 8;
  int k = i & 255;
  Wt[(size_t)c * 256 + k] = (_Float16)W[(size_t)k * ncols + c];
}

// One wave computes one 16x16 output tile of  out = x(16384x256) @ W(256xncols) + bias.
// Wt is f16 transposed [ncols][256] -> B-operand is one contiguous 32B load.
// transpose==0: out f16 row-major [16384][ncols]   (Q=g, K=f, ncols=32)
// transpose==1: out f16 as Vt [B=4][ncols=256][4096] (values, dim-major per batch)
__global__ __launch_bounds__(256) void proj_wmma_kernel(
    const float* __restrict__ x, const _Float16* __restrict__ Wt,
    const float* __restrict__ bias, _Float16* __restrict__ out,
    int ncols, int transpose)
{
  const int lane = threadIdx.x & 31;
  const int wave = threadIdx.x >> 5;
  const int w    = blockIdx.x * 8 + wave;
  const int CT   = ncols >> 4;
  const int rt   = w / CT;
  const int ct   = w % CT;
  const int g    = lane >> 4;
  const int n    = lane & 15;
  const int khiA = g * 16;
  const int khiB = g * 16;

  const float*    xr = x  + (size_t)(rt * 16 + n) * 256;        // A row = lane&15
  const _Float16* wc = Wt + (size_t)(ct * 16 + n) * 256;        // B col = lane&15
  v8f acc = {};

  #pragma unroll 1
  for (int kc = 0; kc < 256; kc += 32) {
    v8f alo = *(const v8f*)(xr + kc + khiA);
    v8f ahi = *(const v8f*)(xr + kc + 8 + khiA);
    v16h a;
    #pragma unroll
    for (int i = 0; i < 8; ++i) { a[i] = (_Float16)alo[i]; a[8 + i] = (_Float16)ahi[i]; }
    v16h bm = *(const v16h*)(wc + kc + khiB);    // K = khiB..khiB+15, contiguous
    acc = wmma_f32_f16(a, bm, acc);
  }

  const float bb = bias[ct * 16 + n];
  if (!transpose) {
    #pragma unroll
    for (int j = 0; j < 8; ++j) {
      int m = 8 * g + j;
      out[(size_t)(rt * 16 + m) * ncols + ct * 16 + n] = (_Float16)(acc[j] + bb);
    }
  } else {
    int keybase = rt * 16 + 8 * g;
    int bidx    = keybase >> 12;
    int key     = keybase & 4095;
    v8h pack;
    #pragma unroll
    for (int j = 0; j < 8; ++j) pack[j] = (_Float16)(acc[j] + bb);
    *(v8h*)(out + (((size_t)(bidx * ncols + ct * 16 + n)) << 12) + key) = pack;
  }
}

#define VROW 40   // padded LDS row stride in halves (32 data + 8 pad)

// Flash attention, block-cooperative: 8 waves share async-staged K/V tiles in LDS.
__global__ __launch_bounds__(256) void flash_attn_kernel(
    const _Float16* __restrict__ Q, const _Float16* __restrict__ K,
    const _Float16* __restrict__ Vt, const float* __restrict__ x,
    const float* __restrict__ gamma_p, float* __restrict__ y)
{
  __shared__ _Float16 sV[2][256 * VROW];   // 2 x 20KB (32-key x 256-dim tile, padded)
  __shared__ _Float16 sK[2][32 * VROW];    // 2 x 2.5KB
  __shared__ _Float16 sP[8][16 * VROW];    // per-wave P staging, padded

  const int tid  = threadIdx.x;
  const int lane = tid & 31;
  const int wave = tid >> 5;
  const int tile = blockIdx.x * 8 + wave;      // 1024 tiles; whole block same batch
  const int b    = tile >> 8;
  const int row0 = (tile & 255) * 16;
  const int g    = lane >> 4;
  const int n    = lane & 15;
  const int khiB = g * 16;
  _Float16* sPw  = &sP[wave][0];

  const _Float16* Kb = K  + (size_t)b * 4096 * 32;
  const _Float16* Vb = Vt + (size_t)b * 256 * 4096;

  // Async stage of one 32-key tile (K: 2KB, V: 16KB) into LDS buffer `buf`.
  auto stage = [&](int buf, int j0) {
    // V: thread tid owns dim row tid -> copies 64B as 4x b128
    const _Float16* gv = Vb + (size_t)tid * 4096 + j0;
    unsigned lv = (unsigned)(uintptr_t)(&sV[buf][tid * VROW]);
    #pragma unroll
    for (int k2 = 0; k2 < 4; ++k2) {
      asm volatile("global_load_async_to_lds_b128 %0, %1, off"
                   :: "v"(lv + k2 * 16),
                      "v"((unsigned long long)(uintptr_t)(gv + k2 * 8))
                   : "memory");
    }
    // K: threads 0..127 copy key rows (32 keys x 64B)
    if (tid < 128) {
      int key = tid >> 2, off = (tid & 3) * 8;
      unsigned lk = (unsigned)(uintptr_t)(&sK[buf][key * VROW + off]);
      asm volatile("global_load_async_to_lds_b128 %0, %1, off"
                   :: "v"(lk),
                      "v"((unsigned long long)(uintptr_t)(Kb + (size_t)(j0 + key) * 32 + off))
                   : "memory");
    }
  };

  // Q tile in A layout
  const _Float16* qr = Q + (size_t)(b * 4096 + row0 + n) * 32;
  v8h q0 = *(const v8h*)(qr + g * 16);
  v8h q1 = *(const v8h*)(qr + 8 + g * 16);
  v16h qa;
  #pragma unroll
  for (int i = 0; i < 8; ++i) { qa[i] = q0[i]; qa[8 + i] = q1[i]; }

  v8f acc[16];
  const v8f zeroc = {};
  #pragma unroll
  for (int c = 0; c < 16; ++c) acc[c] = zeroc;
  float M[8], L[8];
  #pragma unroll
  for (int j = 0; j < 8; ++j) { M[j] = -3.0e38f; L[j] = 0.0f; }

  stage(0, 0);                                  // prologue fetch

  #pragma unroll 1
  for (int it = 0; it < 128; ++it) {
    const int j0  = it * 32;
    const int cur = it & 1;
    asm volatile("s_wait_asynccnt 0" ::: "memory");   // my staged data landed
    __syncthreads();                                  // everyone's data landed
    if (it + 1 < 128) stage(cur ^ 1, j0 + 32);        // overlap next fetch

    // S = Q @ K^T  (two 16x16 tiles, B-operand from LDS)
    const _Float16* kl = &sK[cur][0];
    v16h kb0 = *(const v16h*)(kl + n * VROW + khiB);
    v16h kb1 = *(const v16h*)(kl + (16 + n) * VROW + khiB);
    v8f s0 = wmma_f32_f16(qa, kb0, zeroc);
    v8f s1 = wmma_f32_f16(qa, kb1, zeroc);

    float al[8];
    #pragma unroll
    for (int j = 0; j < 8; ++j) {
      float t = fmaxf(s0[j], s1[j]);
      t = fmaxf(t, __shfl_xor(t, 1));
      t = fmaxf(t, __shfl_xor(t, 2));
      t = fmaxf(t, __shfl_xor(t, 4));
      t = fmaxf(t, __shfl_xor(t, 8));
      float Mn = fmaxf(M[j], t);
      al[j]    = __expf(M[j] - Mn);
      float p0 = __expf(s0[j] - Mn);
      float p1 = __expf(s1[j] - Mn);
      float rs = p0 + p1;
      rs += __shfl_xor(rs, 1);
      rs += __shfl_xor(rs, 2);
      rs += __shfl_xor(rs, 4);
      rs += __shfl_xor(rs, 8);
      L[j] = L[j] * al[j] + rs;
      M[j] = Mn;
      int m = 8 * g + j;
      sPw[m * VROW + n]      = (_Float16)p0;
      sPw[m * VROW + 16 + n] = (_Float16)p1;
    }
    asm volatile("s_wait_dscnt 0" ::: "memory");  // same-wave cross-lane P staging

    v8h p0v = *(const v8h*)(sPw + n * VROW + g * 16);
    v8h p1v = *(const v8h*)(sPw + n * VROW + 8 + g * 16);
    v16h pa;
    #pragma unroll
    for (int i = 0; i < 8; ++i) { pa[i] = p0v[i]; pa[8 + i] = p1v[i]; }

    #pragma unroll
    for (int c = 0; c < 16; ++c) {
      #pragma unroll
      for (int j = 0; j < 8; ++j) acc[c][j] *= al[j];
    }
    const _Float16* vl = &sV[cur][0];
    #pragma unroll
    for (int c = 0; c < 16; ++c) {
      v16h vb = *(const v16h*)(vl + (c * 16 + n) * VROW + khiB);
      acc[c] = wmma_f32_f16(pa, vb, acc[c]);
    }
  }

  const float gm = *gamma_p;
  #pragma unroll
  for (int c = 0; c < 16; ++c) {
    #pragma unroll
    for (int j = 0; j < 8; ++j) {
      int m = 8 * g + j;
      size_t idx = ((size_t)(b * 4096 + row0 + m)) * 256 + c * 16 + n;
      y[idx] = gm * (acc[c][j] / L[j]) + x[idx];
    }
  }
}

extern "C" void kernel_launch(void* const* d_in, const int* in_sizes, int n_in,
                              void* d_out, int out_size, void* d_ws, size_t ws_size,
                              hipStream_t stream) {
  const float* x     = (const float*)d_in[0];
  const float* Wf    = (const float*)d_in[1];  // keys
  const float* bf    = (const float*)d_in[2];
  const float* Wg    = (const float*)d_in[3];  // queries
  const float* bg    = (const float*)d_in[4];
  const float* Wh    = (const float*)d_in[5];  // values
  const float* bh    = (const float*)d_in[6];
  const float* gamma = (const float*)d_in[7];
  float* y = (float*)d_out;

  // Workspace: Q 1MB | K 1MB | Vt 8MB | Wt_g 16KB | Wt_f 16KB | Wt_h 128KB (f16)
  _Float16* Qw  = (_Float16*)d_ws;
  _Float16* Kw  = Qw  + (size_t)16384 * 32;
  _Float16* Vt  = Kw  + (size_t)16384 * 32;
  _Float16* Wtg = Vt  + (size_t)16384 * 256;
  _Float16* Wtf = Wtg + (size_t)32 * 256;
  _Float16* Wth = Wtf + (size_t)32 * 256;

  convert_w_kernel<<<32,  256, 0, stream>>>(Wg, Wtg, 32);
  convert_w_kernel<<<32,  256, 0, stream>>>(Wf, Wtf, 32);
  convert_w_kernel<<<256, 256, 0, stream>>>(Wh, Wth, 256);

  proj_wmma_kernel<<<256,  256, 0, stream>>>(x, Wtg, bg, Qw, 32, 0);   // Q = g
  proj_wmma_kernel<<<256,  256, 0, stream>>>(x, Wtf, bf, Kw, 32, 0);   // K = f
  proj_wmma_kernel<<<2048, 256, 0, stream>>>(x, Wth, bh, Vt, 256, 1);  // V transposed
  flash_attn_kernel<<<128, 256, 0, stream>>>(Qw, Kw, Vt, x, gamma, y);
}
